// AdaptiveNeighborhoodAttention3D_55052890800922
// MI455X (gfx1250) — matrix operand order repaired and checked by reading, not compiled, not run on hardware
//
#include <hip/hip_runtime.h>
#include <hip/hip_bf16.h>

typedef _Float16 f16;
typedef __attribute__((ext_vector_type(8)))  _Float16 v8h;
typedef __attribute__((ext_vector_type(16))) _Float16 v16h;
typedef __attribute__((ext_vector_type(8)))  float    v8f;

#define D_  5
#define H_  64
#define W_  128
#define C_  256
#define NH_ 8
#define HD_ 32
#define L_  (D_*H_*W_)      /* 40960 tokens */

#define TM 64
#define TN 64
#define TK 32
#define PAD 8               /* LDS row pad (halves); row stride 80B keeps 16B alignment */

__device__ __forceinline__ v16h make_v16(v8h lo, v8h hi) {
    return __builtin_shufflevector(lo, hi, 0, 1, 2, 3, 4, 5, 6, 7,
                                           8, 9, 10, 11, 12, 13, 14, 15);
}
__device__ __forceinline__ v8h cvt8(float4 a, float4 b) {
    v8h r;
    r[0] = (f16)a.x; r[1] = (f16)a.y; r[2] = (f16)a.z; r[3] = (f16)a.w;
    r[4] = (f16)b.x; r[5] = (f16)b.y; r[6] = (f16)b.z; r[7] = (f16)b.w;
    return r;
}

// NATTEN window start for k=7, d=2 (nh=3, span = 12)
__device__ __forceinline__ int win_start(int i, int len) {
    int ni = i - 6;
    int imodd = i & 1;
    int b = imodd + ((len - imodd - 1) / 2) * 2;
    int shifted = b - 12;
    if (ni < 0) return imodd;
    if (i + 6 >= len) return shifted;
    return ni;
}

// ---------------------------------------------------------------------------
// Kernel 1: QKV projection GEMM  (M=L_, N=768, K=256), f16 WMMA, fp32 accum.
// ---------------------------------------------------------------------------
__device__ __forceinline__ void store_qkv(float v, int m, int n,
                                          f16* Qb, f16* Kb, f16* Vb, float qs) {
    if (n < 256)      Qb[(size_t)m * 256 + n]         = (f16)(v * qs);
    else if (n < 512) Kb[(size_t)m * 256 + (n - 256)] = (f16)v;
    else              Vb[(size_t)m * 256 + (n - 512)] = (f16)v;
}

__global__ __launch_bounds__(256) void qkv_gemm(const float* __restrict__ X,
                                                const float* __restrict__ Wq,   // [768][256]
                                                const float* __restrict__ bias, // [768]
                                                f16* __restrict__ Qb,
                                                f16* __restrict__ Kb,
                                                f16* __restrict__ Vb) {
    __shared__ f16 As[TM][TK + PAD];   // row-major A tile
    __shared__ f16 Bs[TN][TK + PAD];   // TRANSPOSED B tile: Bs[n][k]
    const int tid  = threadIdx.x;
    const int lane = tid & 31;
    const int wid  = tid >> 5;
    const bool hif = lane >= 16;
    const int  l   = lane & 15;
    const int bm = blockIdx.x * TM;
    const int bn = blockIdx.y * TN;
    const int wm = (wid & 3) * 16;     // 4 M sub-tiles
    const int wn = (wid >> 2) * 32;    // 2 N sub-tiles of 32
    const int sr  = tid >> 2;          // staging row 0..63
    const int sc8 = (tid & 3) * 8;     // staging k-offset {0,8,16,24}
    const int koff = hif ? 8 : 0;      // A-frag K sub-offset
    const int kbo  = hif ? 16 : 0;     // unused; B-frags also use koff

    v8f c0 = {}; v8f c1 = {};
    for (int k0 = 0; k0 < 256; k0 += TK) {
        {   // stage: two float4 loads -> one v8h LDS store each
            const float4* xp = (const float4*)(X + (size_t)(bm + sr) * 256 + k0 + sc8);
            *(v8h*)&As[sr][sc8] = cvt8(xp[0], xp[1]);
            const float4* wp = (const float4*)(Wq + (size_t)(bn + sr) * 256 + k0 + sc8);
            *(v8h*)&Bs[sr][sc8] = cvt8(wp[0], wp[1]);
        }
        if (k0 + TK < 256)  // hint next A tile (global_prefetch_b8)
            __builtin_prefetch(&X[(size_t)(bm + sr) * 256 + (k0 + TK)], 0, 1);
        __syncthreads();

        v16h a  = make_v16(*(const v8h*)&As[wm + l][koff],
                           *(const v8h*)&As[wm + l][16 + koff]);
        v16h b0 = make_v16(*(const v8h*)&Bs[wn + l][koff],
                           *(const v8h*)&Bs[wn + l][16 + koff]);
        v16h b1 = make_v16(*(const v8h*)&Bs[wn + 16 + l][koff],
                           *(const v8h*)&Bs[wn + 16 + l][16 + koff]);
        c0 = __builtin_amdgcn_wmma_f32_16x16x32_f16(false, a, false, b0, (short)0, c0, false, false);
        c1 = __builtin_amdgcn_wmma_f32_16x16x32_f16(false, a, false, b1, (short)0, c1, false, false);
        __syncthreads();
    }
    const float qscale = 0.17677669529663687f;  // 32^-0.5
#pragma unroll
    for (int r = 0; r < 8; ++r) {
        int m = bm + wm + r + (hif ? 8 : 0);
        int n0 = bn + wn + l;
        store_qkv(c0[r] + bias[n0],      m, n0,      Qb, Kb, Vb, qscale);
        store_qkv(c1[r] + bias[n0 + 16], m, n0 + 16, Qb, Kb, Vb, qscale);
    }
}

// ---------------------------------------------------------------------------
// Kernel 2: 3D neighborhood flash-attention.
// block = (z, h, 16-query w-tile); 8 waves = 8 heads. 245 neighbors/query.
// ---------------------------------------------------------------------------
__global__ __launch_bounds__(256) void natten_attn(const f16* __restrict__ Qb,
                                                   const f16* __restrict__ Kb,
                                                   const f16* __restrict__ Vb,
                                                   const float* __restrict__ mask,
                                                   f16* __restrict__ Ob) {
    __shared__ f16 Pls[NH_][16][32 + PAD];   // per-wave P transpose bounce
    __shared__ f16 Vt[NH_][HD_][32 + PAD];   // per-wave transposed V slice [hd][key]
    const int tid  = threadIdx.x;
    const int lane = tid & 31;
    const int nh   = tid >> 5;       // head = wave id
    const bool hif = lane >= 16;
    const int  l   = lane & 15;
    const int w0 = blockIdx.x * 16;
    const int h  = blockIdx.y;
    const int z  = blockIdx.z;
    const int koff = hif ? 8 : 0;    // A-frag K sub-offset
    const int kbo  = hif ? 16 : 0;   // unused; B-frags also use koff

    const int sh  = win_start(h, H_);
    const int wlo = win_start(w0, W_);   // start is monotone in w -> tile min

    // Q A-fragment (two global_load_b128)
    v16h qa;
    {
        const f16* qp = Qb + ((((size_t)z * H_ + h) * W_ + (w0 + l)) * NH_ + nh) * HD_;
        qa = make_v16(*(const v8h*)(qp + koff), *(const v8h*)(qp + 16 + koff));
    }
    // per-accumulator-row query metadata (m = r + 8*hi)
    float maskv[8]; int startw[8];
#pragma unroll
    for (int r = 0; r < 8; ++r) {
        int qw = w0 + r + (hif ? 8 : 0);
        maskv[r]  = mask[((size_t)z * H_ + h) * W_ + qw];
        startw[r] = win_start(qw, W_);
    }

    float mrun[8], lrun[8];
    v8f o0 = {}, o1 = {};
#pragma unroll
    for (int r = 0; r < 8; ++r) { mrun[r] = -1e30f; lrun[r] = 0.0f; }

    for (int zn = 0; zn < D_; ++zn) {
        for (int hj = 0; hj < 7; ++hj) {
            const int hn = sh + hj * 2;
            const f16* kbase = Kb + (((size_t)zn * H_ + hn) * W_ * NH_ + nh) * HD_;
            const f16* vbase = Vb + (((size_t)zn * H_ + hn) * W_ * NH_ + nh) * HD_;

            // ---- stage this slice's V (32 keys x 32 hd) transposed into LDS:
            // lane = key row: 4x global_load_b128 in, b16 transpose stores.
            {
                int wv = wlo + lane;
                if (wv > W_ - 1) wv = W_ - 1;
                const v8h* vp = (const v8h*)(vbase + (size_t)wv * (NH_ * HD_));
                v8h r0 = vp[0], r1 = vp[1], r2 = vp[2], r3 = vp[3];
#pragma unroll
                for (int j = 0; j < 8; ++j) {
                    Vt[nh][j][lane]      = r0[j];
                    Vt[nh][8 + j][lane]  = r1[j];
                    Vt[nh][16 + j][lane] = r2[j];
                    Vt[nh][24 + j][lane] = r3[j];
                }
            }

            // ---- QK: two 16-key groups (each B-frag = two global_load_b128)
            v8f sc[2];
#pragma unroll
            for (int kg = 0; kg < 2; ++kg) {
                int wk = wlo + kg * 16 + l;
                if (wk > W_ - 1) wk = W_ - 1;
                const f16* kp = kbase + (size_t)wk * (NH_ * HD_) + koff;
                v16h kfrag = make_v16(*(const v8h*)kp, *(const v8h*)(kp + 16));
                v8f zz = {};
                sc[kg] = __builtin_amdgcn_wmma_f32_16x16x32_f16(false, qa, false, kfrag,
                                                                (short)0, zz, false, false);
            }
            // ---- validity + logit mask, chunk max
            float p[2][8], cmax[8];
#pragma unroll
            for (int r = 0; r < 8; ++r) cmax[r] = -1e30f;
#pragma unroll
            for (int kg = 0; kg < 2; ++kg) {
                int key_w = wlo + kg * 16 + l;
#pragma unroll
                for (int r = 0; r < 8; ++r) {
                    int dw = key_w - startw[r];
                    bool valid = (key_w < W_) && (dw >= 0) && (dw <= 12) && ((dw & 1) == 0);
                    float s = valid ? sc[kg][r] * maskv[r] : -1e30f;
                    p[kg][r] = s;
                    cmax[r] = fmaxf(cmax[r], s);
                }
            }
#pragma unroll
            for (int off = 1; off < 16; off <<= 1)
#pragma unroll
                for (int r = 0; r < 8; ++r)
                    cmax[r] = fmaxf(cmax[r], __shfl_xor(cmax[r], off, 32));
            // ---- online softmax update
            float fscale[8], csum[8];
#pragma unroll
            for (int r = 0; r < 8; ++r) {
                float mnew = fmaxf(mrun[r], cmax[r]);
                fscale[r] = __expf(mrun[r] - mnew);
                mrun[r] = mnew;
                csum[r] = 0.0f;
            }
#pragma unroll
            for (int kg = 0; kg < 2; ++kg)
#pragma unroll
                for (int r = 0; r < 8; ++r) {
                    float e = __expf(p[kg][r] - mrun[r]);
                    p[kg][r] = e;
                    csum[r] += e;
                }
#pragma unroll
            for (int off = 1; off < 16; off <<= 1)
#pragma unroll
                for (int r = 0; r < 8; ++r) csum[r] += __shfl_xor(csum[r], off, 32);
#pragma unroll
            for (int r = 0; r < 8; ++r) {
                lrun[r] = lrun[r] * fscale[r] + csum[r];
                o0[r] *= fscale[r];
                o1[r] *= fscale[r];
            }
            // ---- P: C-layout -> A-layout via per-wave LDS bounce
#pragma unroll
            for (int kg = 0; kg < 2; ++kg)
#pragma unroll
                for (int r = 0; r < 8; ++r)
                    Pls[nh][r + (hif ? 8 : 0)][kg * 16 + l] = (f16)p[kg][r];
            v16h pa = make_v16(*(const v8h*)&Pls[nh][l][koff],
                               *(const v8h*)&Pls[nh][l][16 + koff]);
            // ---- V B-fragments from transposed LDS slice (2x ds_load_b128 each)
            v16h vb0 = make_v16(*(const v8h*)&Vt[nh][l][koff],
                                *(const v8h*)&Vt[nh][l][16 + koff]);
            v16h vb1 = make_v16(*(const v8h*)&Vt[nh][16 + l][koff],
                                *(const v8h*)&Vt[nh][16 + l][16 + koff]);
            o0 = __builtin_amdgcn_wmma_f32_16x16x32_f16(false, pa, false, vb0, (short)0, o0, false, false);
            o1 = __builtin_amdgcn_wmma_f32_16x16x32_f16(false, pa, false, vb1, (short)0, o1, false, false);
        }
    }
    // ---- finalize: divide by row sum, store f16 in [z][h][w][nh][hd]
#pragma unroll
    for (int r = 0; r < 8; ++r) {
        int qw = w0 + r + (hif ? 8 : 0);
        float inv = 1.0f / lrun[r];
        size_t base = ((((size_t)z * H_ + h) * W_ + qw) * NH_ + nh) * HD_;
        Ob[base + l]      = (f16)(o0[r] * inv);
        Ob[base + 16 + l] = (f16)(o1[r] * inv);
    }
}

// ---------------------------------------------------------------------------
// Kernel 3: output projection GEMM (M=L_, N=256, K=256) + bias, fp32 out.
// ---------------------------------------------------------------------------
__global__ __launch_bounds__(256) void proj_gemm(const f16* __restrict__ A,
                                                 const float* __restrict__ Wp,   // [256][256]
                                                 const float* __restrict__ bias, // [256]
                                                 float* __restrict__ out) {
    __shared__ f16 As[TM][TK + PAD];
    __shared__ f16 Bs[TN][TK + PAD];   // transposed B tile
    const int tid  = threadIdx.x;
    const int lane = tid & 31;
    const int wid  = tid >> 5;
    const bool hif = lane >= 16;
    const int  l   = lane & 15;
    const int bm = blockIdx.x * TM;
    const int bn = blockIdx.y * TN;
    const int wm = (wid & 3) * 16;
    const int wn = (wid >> 2) * 32;
    const int sr  = tid >> 2;
    const int sc8 = (tid & 3) * 8;
    const int koff = hif ? 8 : 0;
    const int kbo  = hif ? 16 : 0;

    v8f c0 = {}; v8f c1 = {};
    for (int k0 = 0; k0 < 256; k0 += TK) {
        {   // A is already f16: one 16B global load per thread
            *(v8h*)&As[sr][sc8] = *(const v8h*)(A + (size_t)(bm + sr) * 256 + k0 + sc8);
            const float4* wp = (const float4*)(Wp + (size_t)(bn + sr) * 256 + k0 + sc8);
            *(v8h*)&Bs[sr][sc8] = cvt8(wp[0], wp[1]);
        }
        __syncthreads();
        v16h a  = make_v16(*(const v8h*)&As[wm + l][koff],
                           *(const v8h*)&As[wm + l][16 + koff]);
        v16h b0 = make_v16(*(const v8h*)&Bs[wn + l][koff],
                           *(const v8h*)&Bs[wn + l][16 + koff]);
        v16h b1 = make_v16(*(const v8h*)&Bs[wn + 16 + l][koff],
                           *(const v8h*)&Bs[wn + 16 + l][16 + koff]);
        c0 = __builtin_amdgcn_wmma_f32_16x16x32_f16(false, a, false, b0, (short)0, c0, false, false);
        c1 = __builtin_amdgcn_wmma_f32_16x16x32_f16(false, a, false, b1, (short)0, c1, false, false);
        __syncthreads();
    }
#pragma unroll
    for (int r = 0; r < 8; ++r) {
        int m = bm + wm + r + (hif ? 8 : 0);
        int n0 = bn + wn + l;
        out[(size_t)m * 256 + n0]      = c0[r] + bias[n0];
        out[(size_t)m * 256 + n0 + 16] = c1[r] + bias[n0 + 16];
    }
}

// ---------------------------------------------------------------------------
extern "C" void kernel_launch(void* const* d_in, const int* in_sizes, int n_in,
                              void* d_out, int out_size, void* d_ws, size_t ws_size,
                              hipStream_t stream) {
    const float* x      = (const float*)d_in[0];
    const float* mask   = (const float*)d_in[1];
    const float* qkv_w  = (const float*)d_in[2];
    const float* qkv_b  = (const float*)d_in[3];
    const float* proj_w = (const float*)d_in[4];
    const float* proj_b = (const float*)d_in[5];
    float* out = (float*)d_out;

    const size_t BUF = (size_t)L_ * C_ * sizeof(f16);  // 20 MB each
    char* ws = (char*)d_ws;
    f16* Qb = (f16*)(ws);
    f16* Kb = (f16*)(ws + BUF);
    f16* Vb = (f16*)(ws + 2 * BUF);
    f16* Ob = (f16*)(ws + 3 * BUF);

    qkv_gemm<<<dim3(L_ / TM, 768 / TN), 256, 0, stream>>>(x, qkv_w, qkv_b, Qb, Kb, Vb);
    natten_attn<<<dim3(W_ / 16, H_, D_), 256, 0, stream>>>(Qb, Kb, Vb, mask, Ob);
    proj_gemm<<<dim3(L_ / TM, C_ / TN), 256, 0, stream>>>(Ob, proj_w, proj_b, out);
}
